// CharRNN_75376676045490
// MI455X (gfx1250) — compile-verified, matrix operand order checked by reading, not run on hardware
//
#include <hip/hip_runtime.h>

// ---------------------------------------------------------------------------
// CharRNN highway-GRU for MI455X (gfx1250), wave32 + WMMA bf16.
// B=128, T=256, R=1024, NU=256, D=3, V=50000.
// ---------------------------------------------------------------------------

#define BB   128
#define TT   256
#define RR   1024
#define NUU  256
#define NWG  32
#define S_ELEMS (BB * RR)   // one state buffer: 128*1024

typedef __attribute__((ext_vector_type(16))) __bf16 bf16x16;
typedef __attribute__((ext_vector_type(8)))  float  floatx8;

__device__ inline floatx8 wmma_bf16(bf16x16 a, bf16x16 b, floatx8 c) {
  // D = A(16x32) * B(32x16) + C(16x16 f32)
  return __builtin_amdgcn_wmma_f32_16x16x32_bf16(
      /*neg_a=*/false, a, /*neg_b=*/false, b,
      /*c_mod=*/(short)0, c, /*reuse_a=*/false, /*reuse_b=*/false);
}

__device__ inline bf16x16 ldb16(const __bf16* p) {
  return *(const bf16x16*)p;   // 32B per lane (2x b128)
}

__device__ inline float sigmoidf_(float x) {
  return 1.0f / (1.0f + __expf(-x));
}

// ---------------------------------------------------------------------------
// Grid-wide sense-reversing barrier (bar[0]=counter, bar[1]=generation)
// ---------------------------------------------------------------------------
__device__ inline void grid_barrier(unsigned* bar) {
  __syncthreads();
  __threadfence();                       // release our state writes
  if (threadIdx.x == 0) {
    volatile unsigned* vgen = bar + 1;
    unsigned g = *vgen;
    unsigned prev = atomicAdd(bar, 1u);
    if (prev == (unsigned)(NWG - 1)) {
      bar[0] = 0;
      __threadfence();
      *vgen = g + 1u;
    } else {
      while (*vgen == g) { __builtin_amdgcn_s_sleep(2); }
    }
  }
  __syncthreads();
  __threadfence();                       // acquire other WGs' state writes
}

// ---------------------------------------------------------------------------
// K=RR reduction over current bf16 state for both gates, 2 column tiles.
// A: rows [rowBase..rowBase+15] of sA.  B: transposed weights [N][K].
// ---------------------------------------------------------------------------
__device__ inline void mm_state(const __bf16* __restrict__ sA,
                                const __bf16* __restrict__ WhT,
                                const __bf16* __restrict__ WtT,
                                int rowBase, int c0, int l16, int halfl,
                                floatx8& h0, floatx8& h1,
                                floatx8& t0, floatx8& t1) {
  const __bf16* aRow = sA  + (size_t)(rowBase + l16) * RR + halfl * 16;
  const __bf16* bh0p = WhT + (size_t)(c0 + l16)      * RR + halfl * 16;
  const __bf16* bh1p = bh0p + 16 * RR;
  const __bf16* bt0p = WtT + (size_t)(c0 + l16)      * RR + halfl * 16;
  const __bf16* bt1p = bt0p + 16 * RR;
#pragma unroll 8
  for (int kk = 0; kk < RR / 32; ++kk) {
    bf16x16 a = ldb16(aRow + kk * 32);
    h0 = wmma_bf16(a, ldb16(bh0p + kk * 32), h0);
    h1 = wmma_bf16(a, ldb16(bh1p + kk * 32), h1);
    t0 = wmma_bf16(a, ldb16(bt0p + kk * 32), t0);
    t1 = wmma_bf16(a, ldb16(bt1p + kk * 32), t1);
  }
}

// ---------------------------------------------------------------------------
// Highway elementwise: s' = (tanh(h)-s)*sigmoid(t) + s, on the 16x16 C layout
// (VGPR j of lane l -> row = rowBase + 8*(l/16) + j, col = cTile + (l&15)).
// Writes fp32 master copy + bf16 mirror.
// ---------------------------------------------------------------------------
__device__ inline void highway_update(const floatx8& h0, const floatx8& h1,
                                      const floatx8& t0, const floatx8& t1,
                                      const float* __restrict__ bh,
                                      const float* __restrict__ bt,
                                      const float* __restrict__ sOld,
                                      float* __restrict__ sNew,
                                      __bf16* __restrict__ sNewBf,
                                      int rowBase, int c0, int l16, int halfl) {
  const int c1 = c0 + 16;
  const float bh0v = bh[c0 + l16], bh1v = bh[c1 + l16];
  const float bt0v = bt[c0 + l16], bt1v = bt[c1 + l16];
  const int r0 = rowBase + halfl * 8;
#pragma unroll
  for (int j = 0; j < 8; ++j) {
    const int r = r0 + j;
    {
      const size_t o = (size_t)r * RR + c0 + l16;
      const float so = sOld[o];
      const float hv = tanhf(h0[j] + bh0v);
      const float gv = sigmoidf_(t0[j] + bt0v);
      const float sn = (hv - so) * gv + so;
      sNew[o] = sn; sNewBf[o] = (__bf16)sn;
    }
    {
      const size_t o = (size_t)r * RR + c1 + l16;
      const float so = sOld[o];
      const float hv = tanhf(h1[j] + bh1v);
      const float gv = sigmoidf_(t1[j] + bt1v);
      const float sn = (hv - so) * gv + so;
      sNew[o] = sn; sNewBf[o] = (__bf16)sn;
    }
  }
}

// ---------------------------------------------------------------------------
// Prep kernels
// ---------------------------------------------------------------------------
__global__ void init_state_kernel(float* s32, __bf16* sbf, unsigned* bar) {
  const size_t i = (size_t)blockIdx.x * blockDim.x + threadIdx.x;
  if (i < 2u * S_ELEMS) { s32[i] = 0.0f; sbf[i] = (__bf16)0.0f; }
  if (i == 0) { bar[0] = 0u; bar[1] = 0u; }
}

// dst[n*K + k] = (bf16) src[k*N + n]   (transpose + convert)
__global__ void transpose_bf16_kernel(__bf16* __restrict__ dst,
                                      const float* __restrict__ src,
                                      int K, int N) {
  const int idx = blockIdx.x * blockDim.x + threadIdx.x;
  if (idx < K * N) {
    const int n = idx / K;
    const int k = idx - n * K;
    dst[idx] = (__bf16)src[(size_t)k * N + n];
  }
}

// ---------------------------------------------------------------------------
// Persistent recurrent kernel: 32 WGs x 256 threads (8 wave32s).
// WG owns columns [32*wg, 32*wg+32); wave owns rows [16*wave, 16*wave+16).
// ---------------------------------------------------------------------------
__global__ void __launch_bounds__(256, 1)
rnn_persistent(const int*   __restrict__ input,
               const float* __restrict__ emb,
               const __bf16* __restrict__ WhT0x, const __bf16* __restrict__ WtT0x,
               const __bf16* __restrict__ WhT0s, const __bf16* __restrict__ WtT0s,
               const __bf16* __restrict__ WhhT,  const __bf16* __restrict__ WthT,
               const __bf16* __restrict__ WpT,
               const float* __restrict__ bh0,  const float* __restrict__ bt0,
               const float* __restrict__ bhh,  const float* __restrict__ bth,
               const float* __restrict__ bp,
               float* __restrict__ s32, __bf16* __restrict__ sbf,
               float* __restrict__ out, unsigned* bar) {
  __shared__ __bf16 xsh[BB * NUU];             // 64 KB: x_t as bf16

  const int tid     = threadIdx.x;
  const int wave    = tid >> 5;
  const int lane    = tid & 31;
  const int l16     = lane & 15;
  const int halfl   = lane >> 4;
  const int rowBase = wave * 16;
  const int colBase = blockIdx.x * 32;

  int p = 0;                                   // state buffer parity
  for (int t = 0; t < TT; ++t) {
    // ---- stage x_t = embedding[input[:, t]] into LDS (bf16) ----
    {
      const int row  = tid >> 1;
      const int half = tid & 1;
      const int vidx = input[row * TT + t];
      const float4* src = (const float4*)(emb + (size_t)vidx * NUU + half * 128);
      __bf16* dst = xsh + row * NUU + half * 128;
#pragma unroll 8
      for (int i = 0; i < 32; ++i) {
        const float4 v = src[i];
        dst[i * 4 + 0] = (__bf16)v.x;
        dst[i * 4 + 1] = (__bf16)v.y;
        dst[i * 4 + 2] = (__bf16)v.z;
        dst[i * 4 + 3] = (__bf16)v.w;
      }
    }
    __syncthreads();

    // ---- layer 0: gates over [x_t, s] ----
    floatx8 h0 = {0,0,0,0,0,0,0,0}, h1 = {0,0,0,0,0,0,0,0};
    floatx8 g0 = {0,0,0,0,0,0,0,0}, g1 = {0,0,0,0,0,0,0,0};
    {
      // x-part: K = NUU, A from LDS, B from transposed input weights
      const __bf16* aRow = xsh + (rowBase + l16) * NUU + halfl * 16;
      const __bf16* bhp  = WhT0x + (size_t)(colBase + l16) * NUU + halfl * 16;
      const __bf16* btp  = WtT0x + (size_t)(colBase + l16) * NUU + halfl * 16;
#pragma unroll
      for (int kk = 0; kk < NUU / 32; ++kk) {
        bf16x16 a = ldb16(aRow + kk * 32);
        h0 = wmma_bf16(a, ldb16(bhp + kk * 32), h0);
        h1 = wmma_bf16(a, ldb16(bhp + 16 * NUU + kk * 32), h1);
        g0 = wmma_bf16(a, ldb16(btp + kk * 32), g0);
        g1 = wmma_bf16(a, ldb16(btp + 16 * NUU + kk * 32), g1);
      }
    }
    mm_state(sbf + (size_t)p * S_ELEMS, WhT0s, WtT0s,
             rowBase, colBase, l16, halfl, h0, h1, g0, g1);
    highway_update(h0, h1, g0, g1, bh0, bt0,
                   s32 + (size_t)p * S_ELEMS,
                   s32 + (size_t)(1 - p) * S_ELEMS,
                   sbf + (size_t)(1 - p) * S_ELEMS,
                   rowBase, colBase, l16, halfl);
    grid_barrier(bar); p ^= 1;

    // ---- layers 1..D-1: gates over s only ----
#pragma unroll
    for (int l = 0; l < 2; ++l) {
      floatx8 a0 = {0,0,0,0,0,0,0,0}, a1 = {0,0,0,0,0,0,0,0};
      floatx8 b0 = {0,0,0,0,0,0,0,0}, b1 = {0,0,0,0,0,0,0,0};
      mm_state(sbf + (size_t)p * S_ELEMS,
               WhhT + (size_t)l * RR * RR, WthT + (size_t)l * RR * RR,
               rowBase, colBase, l16, halfl, a0, a1, b0, b1);
      highway_update(a0, a1, b0, b1, bhh + l * RR, bth + l * RR,
                     s32 + (size_t)p * S_ELEMS,
                     s32 + (size_t)(1 - p) * S_ELEMS,
                     sbf + (size_t)(1 - p) * S_ELEMS,
                     rowBase, colBase, l16, halfl);
      grid_barrier(bar); p ^= 1;
    }

    // ---- output projection out_t = s @ Wp + bp, by WGs 0..7 ----
    // Overlaps with other WGs' layer-0 of step t+1: they only READ sbf[p],
    // and the first WRITE to sbf[p] happens after the next grid barrier.
    if (blockIdx.x < 8) {
      const int prow = blockIdx.x * 16;
      const int pc0  = wave * 32;
      floatx8 o0 = {0,0,0,0,0,0,0,0}, o1 = {0,0,0,0,0,0,0,0};
      const __bf16* sA = sbf + (size_t)p * S_ELEMS
                             + (size_t)(prow + l16) * RR + halfl * 16;
      const __bf16* w0 = WpT + (size_t)(pc0 + l16) * RR + halfl * 16;
      const __bf16* w1 = w0 + 16 * RR;
#pragma unroll 8
      for (int kk = 0; kk < RR / 32; ++kk) {
        bf16x16 a = ldb16(sA + kk * 32);
        o0 = wmma_bf16(a, ldb16(w0 + kk * 32), o0);
        o1 = wmma_bf16(a, ldb16(w1 + kk * 32), o1);
      }
      const float bp0 = bp[pc0 + l16];
      const float bp1 = bp[pc0 + 16 + l16];
      const int r0 = prow + halfl * 8;
#pragma unroll
      for (int j = 0; j < 8; ++j) {
        const int r = r0 + j;
        out[((size_t)r * TT + t) * NUU + pc0 + l16]      = o0[j] + bp0;
        out[((size_t)r * TT + t) * NUU + pc0 + 16 + l16] = o1[j] + bp1;
      }
    }
  }
}

// ---------------------------------------------------------------------------
// Host launcher
// ---------------------------------------------------------------------------
extern "C" void kernel_launch(void* const* d_in, const int* in_sizes, int n_in,
                              void* d_out, int out_size, void* d_ws, size_t ws_size,
                              hipStream_t stream) {
  (void)in_sizes; (void)n_in; (void)out_size; (void)ws_size;

  const int*   input = (const int*)  d_in[0];
  const float* emb   = (const float*)d_in[1];
  const float* Wh0x  = (const float*)d_in[2];
  const float* Wh0s  = (const float*)d_in[3];
  const float* bh0   = (const float*)d_in[4];
  const float* Wt0x  = (const float*)d_in[5];
  const float* Wt0s  = (const float*)d_in[6];
  const float* bt0   = (const float*)d_in[7];
  const float* Whh   = (const float*)d_in[8];
  const float* bhh   = (const float*)d_in[9];
  const float* Wth   = (const float*)d_in[10];
  const float* bth   = (const float*)d_in[11];
  const float* Wp    = (const float*)d_in[12];
  const float* bp    = (const float*)d_in[13];
  float* out = (float*)d_out;

  // ---- workspace layout ----
  char* ws = (char*)d_ws;
  unsigned* bar = (unsigned*)ws;                                   // 256 B
  float*  s32 = (float*) (ws + 256);                               // 2*128K f32
  __bf16* sbf = (__bf16*)(ws + 256 + 2ull * S_ELEMS * 4);          // 2*128K bf16
  __bf16* wts = (__bf16*)(ws + 256 + 2ull * S_ELEMS * 4 + 2ull * S_ELEMS * 2);
  __bf16* WhT0x = wts;                     // [1024][256]
  __bf16* WtT0x = WhT0x + 262144;          // [1024][256]
  __bf16* WhT0s = WtT0x + 262144;          // [1024][1024]
  __bf16* WtT0s = WhT0s + 1048576;         // [1024][1024]
  __bf16* WhhT  = WtT0s + 1048576;         // 2 x [1024][1024]
  __bf16* WthT  = WhhT  + 2 * 1048576;     // 2 x [1024][1024]
  __bf16* WpT   = WthT  + 2 * 1048576;     // [256][1024]

  // ---- prep: zero state + barrier, transpose-convert weights to bf16 ----
  {
    const int n = 2 * S_ELEMS;
    init_state_kernel<<<(n + 255) / 256, 256, 0, stream>>>(s32, sbf, bar);
  }
  {
    const int nSmall = 256 * 1024, nBig = 1024 * 1024;
    const int gS = (nSmall + 255) / 256, gB = (nBig + 255) / 256;
    transpose_bf16_kernel<<<gS, 256, 0, stream>>>(WhT0x, Wh0x, 256, 1024);
    transpose_bf16_kernel<<<gS, 256, 0, stream>>>(WtT0x, Wt0x, 256, 1024);
    transpose_bf16_kernel<<<gB, 256, 0, stream>>>(WhT0s, Wh0s, 1024, 1024);
    transpose_bf16_kernel<<<gB, 256, 0, stream>>>(WtT0s, Wt0s, 1024, 1024);
    transpose_bf16_kernel<<<gB, 256, 0, stream>>>(WhhT,           Whh,            1024, 1024);
    transpose_bf16_kernel<<<gB, 256, 0, stream>>>(WhhT + 1048576, Whh + 1048576,  1024, 1024);
    transpose_bf16_kernel<<<gB, 256, 0, stream>>>(WthT,           Wth,            1024, 1024);
    transpose_bf16_kernel<<<gB, 256, 0, stream>>>(WthT + 1048576, Wth + 1048576,  1024, 1024);
    transpose_bf16_kernel<<<gS, 256, 0, stream>>>(WpT, Wp, 1024, 256);
  }

  // ---- persistent recurrent kernel ----
  rnn_persistent<<<NWG, 256, 0, stream>>>(
      input, emb,
      WhT0x, WtT0x, WhT0s, WtT0s, WhhT, WthT, WpT,
      bh0, bt0, bhh, bth, bp,
      s32, sbf, out, bar);
}
